// QuestionSpecificMLP_19335942766765
// MI455X (gfx1250) — compile-verified, matrix operand order checked by reading, not run on hardware
//
#include <hip/hip_runtime.h>
#include <hip/hip_bf16.h>

// ---------------- types ----------------
typedef __attribute__((ext_vector_type(16))) __bf16         v16bf;
typedef __attribute__((ext_vector_type(16))) unsigned short v16u;
typedef __attribute__((ext_vector_type(8)))  unsigned short v8u;
typedef __attribute__((ext_vector_type(8)))  float          v8f;

#define B_SZ 32768
#define Q_SZ 512
#define H_SZ 16
#define D_SZ 192

// ---------------- bf16 helpers (bit-exact, no __bf16 arithmetic) ----------------
__device__ __forceinline__ unsigned short f2bf(float f) {
    unsigned u = __float_as_uint(f);
    unsigned r = (u + 0x7FFFu + ((u >> 16) & 1u)) >> 16;   // RNE
    return (unsigned short)r;
}
__device__ __forceinline__ float bf2f(unsigned short s) {
    return __uint_as_float(((unsigned)s) << 16);
}
__device__ __forceinline__ float gelu_tanh(float x) {
    const float k0 = 0.7978845608028654f;   // sqrt(2/pi)
    const float k1 = 0.044715f;
    float t = tanhf(k0 * (x + k1 * x * x * x));
    return 0.5f * x * (1.0f + t);
}

// Load a 16-half WMMA A/B fragment for this lane (32-bit element offset so the
// backend can use SADDR(base64) + VGPR(offset32) addressing).
// 16-bit operand layout (ISA 7.12.2): lanes 0-15 kBase=0, lanes 16-31 kBase=8;
// halves 0..7  = K kBase..kBase+7, halves 8..15 = K kBase+16..kBase+23.
__device__ __forceinline__ v16u load_frag_u(const unsigned short* base, unsigned eoff) {
    v8u lo = *(const v8u*)(base + eoff);
    v8u hi = *(const v8u*)(base + eoff + 16);
    v16u r;
#pragma unroll
    for (int i = 0; i < 8; ++i) { r[i] = lo[i]; r[i + 8] = hi[i]; }
    return r;
}

// ---------------- weight transpose + bf16 convert: Wt[n][k] = W[k][n] ----------------
__global__ void qsmlp_transpose_w(const float* __restrict__ W,
                                  unsigned short* __restrict__ Wt,
                                  int K, int N) {
    long i = (long)blockIdx.x * blockDim.x + threadIdx.x;
    if (i >= (long)K * N) return;
    int n = (int)(i / K);
    int k = (int)(i % K);
    Wt[i] = f2bf(W[(long)k * N + n]);
}

// ---------------- build combined input: [B, 800] bf16 = cat(x, corr_emb[is_correct]) ----------------
__global__ void qsmlp_build_input(const float* __restrict__ x,
                                  const int* __restrict__ isc,
                                  const float* __restrict__ cemb,
                                  unsigned short* __restrict__ out) {
    long i = (long)blockIdx.x * blockDim.x + threadIdx.x;
    if (i >= (long)B_SZ * 800) return;
    int b = (int)(i / 800);
    int c = (int)(i % 800);
    float v = (c < 768) ? x[(long)b * 768 + c]
                        : cemb[isc[b] * 32 + (c - 768)];
    out[i] = f2bf(v);
}

// ---------------- fused GEMM (bf16 WMMA) + bias + LayerNorm + GELU ----------------
// Block: WAVES waves, 16 rows x N cols. Exact partition (WAVES*T == N/16):
// wave w owns tiles [w*T, (w+1)*T). Small T keeps acc+fragments in registers
// with no copies; k-loop unrolled x2 so consecutive steps use disjoint
// registers (double-buffering without explicit movs).
template <int K, int N, int WAVES, int T>
__global__ __launch_bounds__(WAVES * 32)
void qsmlp_gemm_ln_gelu(const unsigned short* __restrict__ A,   // [B][K] bf16
                        const unsigned short* __restrict__ Wt,  // [N][K] bf16
                        const float* __restrict__ bias,         // [N]
                        const float* __restrict__ g,            // [N]
                        const float* __restrict__ be,           // [N]
                        unsigned short* __restrict__ out)       // [B][N] bf16
{
    constexpr int NT = N / 16;
    static_assert(WAVES * T == NT, "exact tile partition required");
    constexpr int THREADS = WAVES * 32;
    __shared__ float ldsC[16][N];

    const int tid   = threadIdx.x;
    const int wave  = tid >> 5;
    const int lane  = tid & 31;
    const int lo    = lane & 15;
    const int hi    = lane >> 4;
    const int kBase = hi ? 8 : 0;

    // 32-bit element offsets (all buffers < 4GB)
    const unsigned aBase = ((unsigned)blockIdx.x * 16 + lo) * K + kBase;
    unsigned bOff[T];
#pragma unroll
    for (int t = 0; t < T; ++t)
        bOff[t] = (unsigned)((wave * T + t) * 16 + lo) * K + kBase;

    v8f acc[T];
#pragma unroll
    for (int t = 0; t < T; ++t) acc[t] = (v8f){0.f,0.f,0.f,0.f,0.f,0.f,0.f,0.f};

#pragma unroll 2
    for (int k0 = 0; k0 < K; k0 += 32) {
        v16u aF = load_frag_u(A, aBase + k0);
        v16u bF[T];
#pragma unroll
        for (int t = 0; t < T; ++t) bF[t] = load_frag_u(Wt, bOff[t] + k0);
        __builtin_prefetch(A + aBase + k0 + 64, 0, 3);

        const v16bf aC = __builtin_bit_cast(v16bf, aF);
#pragma unroll
        for (int t = 0; t < T; ++t)
            acc[t] = __builtin_amdgcn_wmma_f32_16x16x32_bf16(
                false, aC, false, __builtin_bit_cast(v16bf, bF[t]),
                (short)0, acc[t], false, false);
    }

    // C/D layout: VGPR v -> row (v + 8*hi), col = lane&15 within tile.
#pragma unroll
    for (int t = 0; t < T; ++t) {
        const int col  = (wave * T + t) * 16 + lo;
        const float bc = bias[col];
#pragma unroll
        for (int v = 0; v < 8; ++v)
            ldsC[v + 8 * hi][col] = acc[t][v] + bc;
    }
    __syncthreads();

    // LayerNorm + GELU: 16 threads per row (aligned 16-lane shfl segments).
    const int j = tid & 15;
    for (int r = tid >> 4; r < 16; r += THREADS / 16) {
        float s = 0.f, s2 = 0.f;
#pragma unroll
        for (int c = j; c < N; c += 16) { float v = ldsC[r][c]; s += v; s2 += v * v; }
#pragma unroll
        for (int off = 8; off >= 1; off >>= 1) {
            s  += __shfl_xor(s,  off, 16);
            s2 += __shfl_xor(s2, off, 16);
        }
        const float inv = 1.0f / (float)N;
        const float mu  = s * inv;
        const float rs  = rsqrtf(s2 * inv - mu * mu + 1e-5f);

        unsigned short* oRow = out + ((long)blockIdx.x * 16 + r) * N;
#pragma unroll
        for (int c = j; c < N; c += 16) {
            float v = (ldsC[r][c] - mu) * rs * g[c] + be[c];
            oRow[c] = f2bf(gelu_tanh(v));
        }
    }
}

// ---------------- per-(question,correctness) head ----------------
// Block = 256 threads = 16 samples x 16 outputs. The 16 h-vectors (6KB) are
// staged in LDS with one coalesced cooperative load; head_W reads stay
// lane-coalesced (16 consecutive threads -> 64B per d).
__global__ __launch_bounds__(256)
void qsmlp_head(const unsigned short* __restrict__ h,   // [B][192] bf16
                const int* __restrict__ qid,
                const int* __restrict__ isc,
                const float* __restrict__ hW,            // [2][Q][192][16]
                const float* __restrict__ hb,            // [2][Q][16]
                float* __restrict__ out)                 // [B][16]
{
    __shared__ unsigned short lh[16 * D_SZ];             // 6 KB
    const int tid = threadIdx.x;
    const long b0 = (long)blockIdx.x * 16;

    // 16*192 halves = 1536 dwords; 6 dwords per thread, coalesced.
    const unsigned* src = (const unsigned*)(h + b0 * D_SZ);
    unsigned* dst = (unsigned*)lh;
#pragma unroll
    for (int i = tid; i < 16 * D_SZ / 2; i += 256) dst[i] = src[i];
    __syncthreads();

    const int lb = tid >> 4;          // local sample
    const int hh = tid & 15;          // head output
    const long b = b0 + lb;
    const long base = (long)isc[b] * Q_SZ + (long)qid[b];
    const float* W = hW + base * (D_SZ * H_SZ) + hh;
    const unsigned short* hr = lh + lb * D_SZ;

    float acc = hb[base * H_SZ + hh];
#pragma unroll 8
    for (int d = 0; d < D_SZ; ++d)
        acc = fmaf(bf2f(hr[d]), W[(long)d * H_SZ], acc);
    out[b * H_SZ + hh] = acc;
}

// ---------------- launch ----------------
extern "C" void kernel_launch(void* const* d_in, const int* in_sizes, int n_in,
                              void* d_out, int out_size, void* d_ws, size_t ws_size,
                              hipStream_t stream) {
    (void)in_sizes; (void)n_in; (void)out_size; (void)ws_size;

    const float* x    = (const float*)d_in[0];
    const int*   qid  = (const int*)d_in[1];
    const int*   isc  = (const int*)d_in[2];
    const float* cemb = (const float*)d_in[3];
    const float* W1   = (const float*)d_in[4];
    const float* W2   = (const float*)d_in[5];
    const float* W3   = (const float*)d_in[6];
    const float* W4   = (const float*)d_in[7];
    const float* b1   = (const float*)d_in[8];
    const float* b2   = (const float*)d_in[9];
    const float* b3   = (const float*)d_in[10];
    const float* b4   = (const float*)d_in[11];
    const float* g1   = (const float*)d_in[12];
    const float* g2   = (const float*)d_in[13];
    const float* g3   = (const float*)d_in[14];
    const float* g4   = (const float*)d_in[15];
    const float* be1  = (const float*)d_in[16];
    const float* be2  = (const float*)d_in[17];
    const float* be3  = (const float*)d_in[18];
    const float* be4  = (const float*)d_in[19];
    const float* hW   = (const float*)d_in[20];
    const float* hb   = (const float*)d_in[21];
    float* out = (float*)d_out;

    // workspace layout (halves)
    unsigned short* wt1 = (unsigned short*)d_ws;            // 1024x800
    unsigned short* wt2 = wt1 + 1024 * 800;                 // 512x1024
    unsigned short* wt3 = wt2 + 512 * 1024;                 // 256x512
    unsigned short* wt4 = wt3 + 256 * 512;                  // 192x256
    unsigned short* buf0 = wt4 + 192 * 256;                 // B x 800 (also H2, H4)
    unsigned short* buf1 = buf0 + (long)B_SZ * 800;         // B x 1024 (also H3)

    const int TPB = 256;
    // weight conversion + transpose
    qsmlp_transpose_w<<<(800 * 1024 + TPB - 1) / TPB, TPB, 0, stream>>>(W1, wt1, 800, 1024);
    qsmlp_transpose_w<<<(1024 * 512 + TPB - 1) / TPB, TPB, 0, stream>>>(W2, wt2, 1024, 512);
    qsmlp_transpose_w<<<(512 * 256 + TPB - 1) / TPB, TPB, 0, stream>>>(W3, wt3, 512, 256);
    qsmlp_transpose_w<<<(256 * 192 + TPB - 1) / TPB, TPB, 0, stream>>>(W4, wt4, 256, 192);

    // combined input
    long nIn = (long)B_SZ * 800;
    qsmlp_build_input<<<(unsigned)((nIn + TPB - 1) / TPB), TPB, 0, stream>>>(x, isc, cemb, buf0);

    // trunk: Linear -> LN -> GELU, ping-pong buffers
    // 512-thread (16-wave) blocks with small T keep register pressure low.
    const unsigned gBlocks = B_SZ / 16;
    qsmlp_gemm_ln_gelu< 800, 1024, 16, 4><<<gBlocks, 512, 0, stream>>>(buf0, wt1, b1, g1, be1, buf1);
    qsmlp_gemm_ln_gelu<1024,  512, 16, 2><<<gBlocks, 512, 0, stream>>>(buf1, wt2, b2, g2, be2, buf0);
    qsmlp_gemm_ln_gelu< 512,  256, 16, 1><<<gBlocks, 512, 0, stream>>>(buf0, wt3, b3, g3, be3, buf1);
    qsmlp_gemm_ln_gelu< 256,  192, 12, 1><<<gBlocks, 384, 0, stream>>>(buf1, wt4, b4, g4, be4, buf0);

    // gathered heads
    qsmlp_head<<<(B_SZ * H_SZ) / TPB, TPB, 0, stream>>>(buf0, qid, isc, hW, hb, out);
}
